// EncoderLayer_8306466750635
// MI455X (gfx1250) — matrix operand order repaired and checked by reading, not compiled, not run on hardware
//
#include <hip/hip_runtime.h>

// ---------------------------------------------------------------------------
// Transformer encoder layer for gfx1250 (MI455X), mixed precision:
//   * all GEMM operands stored f16 in HBM (converted once), f32 accumulate
//   * v_wmma_f32_16x16x32_f16, 64x128 (or 64x64) block tiles, 4 (2) WMMA/wave/k-step
//   * A tiles staged via global_load_async_to_lds_b128
//   * B LDS layout is fragment-major: each lane's WMMA B operand = 2x ds_load_b128
// ---------------------------------------------------------------------------

typedef __attribute__((ext_vector_type(16))) _Float16 v16h;
typedef __attribute__((ext_vector_type(8)))  float    v8f;
typedef __attribute__((ext_vector_type(4)))  int      v4i;

union Frag16 { v16h v; unsigned int u[8]; };

__device__ __forceinline__ unsigned int pack2f(float a, float b) {
    union { _Float16 h[2]; unsigned int u; } t;
    t.h[0] = (_Float16)a; t.h[1] = (_Float16)b; return t.u;
}
__device__ __forceinline__ unsigned int pack2h(_Float16 a, _Float16 b) {
    union { _Float16 h[2]; unsigned int u; } t;
    t.h[0] = a; t.h[1] = b; return t.u;
}

#if defined(__HIP_DEVICE_COMPILE__) && __has_builtin(__builtin_amdgcn_global_load_async_to_lds_b128)
#define HAVE_ASYNC_LDS 1
typedef __attribute__((address_space(1))) v4i* async_g_p;   // global int4*
typedef __attribute__((address_space(3))) v4i* async_l_p;   // LDS int4*
#else
#define HAVE_ASYNC_LDS 0
#endif

__device__ __forceinline__ void wait_async0() {
#if defined(__HIP_DEVICE_COMPILE__)
#if __has_builtin(__builtin_amdgcn_s_wait_asynccnt)
    __builtin_amdgcn_s_wait_asynccnt(0);
#elif HAVE_ASYNC_LDS
    asm volatile("s_wait_asynccnt 0x0" ::: "memory");
#endif
#endif
}

// ---------------------------------------------------------------------------
// Batched GEMM: C = act( scale * (A @ B[^T]) + bias ),  A: f16|f32, B: f16.
// Block tile 64 x BNT x 32; 8 wave32 waves as 4(M) x 2(N); wave tile 16 x BNT/2.
// Batch z offset = (z % bdiv) * s1 + (z / bdiv) * s2 per operand.
// All M,N,K are exact multiples of the tile in every call below.
//
// LDS B layout (fragment-major), u32 units:
//   u32 for k-pair kr (0..15) at column n lives at
//       (kr>>3)*(BNT*8) + n*8 + (kr&7)
//   so a lane's 8 B-fragment dwords (kr = 8*hi + 0..7, fixed n) are contiguous.
// ---------------------------------------------------------------------------
template <typename TA, int BNT>
__global__ __launch_bounds__(256) void gemm_wmma_kernel(
    const TA* __restrict__ A, int lda, long long sA1, long long sA2,
    const _Float16* __restrict__ B, int ldb, long long sB1, long long sB2, int transB,
    float* __restrict__ Cf, _Float16* __restrict__ Ch, int ldc, long long sC1, long long sC2,
    const float* __restrict__ bias, float scale, int relu, int bdiv,
    int M, int N, int K)
{
    constexpr int NSUB = BNT / 32;                 // accumulators per wave (2 or 4)
    __shared__ _Float16 lds_a[64 * 32];            // row-major [64][32]
    __shared__ _Float16 lds_b[32 * BNT];           // fragment-major (see above)

    const int tid   = threadIdx.x;
    const int lane  = tid & 31;
    const int wave  = tid >> 5;
    const int waveM = wave & 3;                    // 4 along M
    const int waveN = wave >> 2;                   // 2 along N

    const int z = blockIdx.z;
    const TA*       Ab = A + (long long)(z % bdiv) * sA1 + (long long)(z / bdiv) * sA2;
    const _Float16* Bb = B + (long long)(z % bdiv) * sB1 + (long long)(z / bdiv) * sB2;
    const long long offC = (long long)(z % bdiv) * sC1 + (long long)(z / bdiv) * sC2;

    const int m0 = blockIdx.y * 64;
    const int n0 = blockIdx.x * BNT;

    v8f acc[NSUB] = {};

    const int nlane = lane & 15;
    const int hi    = (lane >> 4) & 1;
    const int mrow  = waveM * 16 + nlane;

    for (int k0 = 0; k0 < K; k0 += 32) {
        // ---------------- stage A tile (64 x 32 halves) ----------------
        if constexpr (sizeof(TA) == 2) {
            const int r   = tid >> 2;
            const int seg = (tid & 3) * 8;         // halves
            const _Float16* src = (const _Float16*)Ab + (long long)(m0 + r) * lda + k0 + seg;
            _Float16* dst = &lds_a[r * 32 + seg];
#if HAVE_ASYNC_LDS
            __builtin_amdgcn_global_load_async_to_lds_b128(
                (async_g_p)(void*)src, (async_l_p)(void*)dst, 0, 0);
#else
            *(uint4*)dst = *(const uint4*)src;
#endif
            if (k0 + 32 < K) __builtin_prefetch(src + 32, 0, 3);
        } else {
            const int r = tid >> 2;
            const int c = (tid & 3) * 8;
            const float4* s4 = (const float4*)((const float*)Ab + (long long)(m0 + r) * lda + k0 + c);
            float4 p0 = s4[0], p1 = s4[1];
            unsigned int* dst = (unsigned int*)&lds_a[r * 32 + c];
            dst[0] = pack2f(p0.x, p0.y);
            dst[1] = pack2f(p0.z, p0.w);
            dst[2] = pack2f(p1.x, p1.y);
            dst[3] = pack2f(p1.z, p1.w);
            if (k0 + 32 < K) __builtin_prefetch((const float*)s4 + 32, 0, 3);
        }

        // ---------------- stage B tile (32 x BNT halves, fragment-major) ----------------
        unsigned int* lb32 = (unsigned int*)lds_b;
        if (!transB) {
            constexpr int CW = BNT / 16;           // columns per thread (8 or 4)
            const int kp = tid >> 4;               // 0..15 (k-pair)
            const int ng = (tid & 15) * CW;
            const _Float16* r0 = Bb + (long long)(k0 + 2 * kp) * ldb + (n0 + ng);
            const _Float16* r1 = r0 + ldb;
            union { uint4 q4; uint2 q2; _Float16 h[8]; } ua, ub;
            if constexpr (CW == 8) { ua.q4 = *(const uint4*)r0; ub.q4 = *(const uint4*)r1; }
            else                   { ua.q2 = *(const uint2*)r0; ub.q2 = *(const uint2*)r1; }
            const int basei = (kp >> 3) * (BNT * 8) + (kp & 7);
            #pragma unroll
            for (int j = 0; j < CW; j++) lb32[basei + (ng + j) * 8] = pack2h(ua.h[j], ub.h[j]);
            if (k0 + 32 < K) __builtin_prefetch(r0 + 32LL * ldb, 0, 3);
        } else {
            // B[k][n] = Bglob[n][k]: k-pairs are contiguous u32 in global memory,
            // and stay contiguous in the fragment-major LDS layout.
            constexpr int TPC = 256 / BNT;         // threads per column (2 or 4)
            constexpr int CW  = 16 / TPC;          // k-pairs per thread (8 or 4)
            const int n   = tid / TPC;
            const int kpg = (tid % TPC) * CW;
            const _Float16* src = Bb + (long long)(n0 + n) * ldb + k0 + 2 * kpg;
            union { uint4 q[2]; unsigned int u[8]; } t;
            t.q[0] = *(const uint4*)src;
            if constexpr (CW == 8) t.q[1] = *(const uint4*)(src + 8);
            const int basei = (kpg >> 3) * (BNT * 8) + n * 8 + (kpg & 7);
            #pragma unroll
            for (int j = 0; j < CW; j++) lb32[basei + j] = t.u[j];
            if (k0 + 32 < K) __builtin_prefetch(src + 32, 0, 3);
        }

        if constexpr (sizeof(TA) == 2) wait_async0();
        __syncthreads();

        // ---------------- fragments + WMMA ----------------
        Frag16 a;
        const unsigned int* la = (const unsigned int*)lds_a;
        #pragma unroll
        for (int v = 0; v < 8; v++) {
            // A 16x32 f16: lanes0-15 hold K 0..15; lanes16-31 hold K 16..31
            a.u[v] = la[mrow * 16 + 8 * hi + v];
        }
        const unsigned int* lb = (const unsigned int*)lds_b;
        #pragma unroll
        for (int sub = 0; sub < NSUB; sub++) {
            Frag16 bf;
            const int col = waveN * (BNT / 2) + sub * 16 + nlane;
            const int fb  = hi * (BNT * 8) + col * 8;
            #pragma unroll
            for (int v = 0; v < 8; v++) bf.u[v] = lb[fb + v];
            acc[sub] = __builtin_amdgcn_wmma_f32_16x16x32_f16(false, a.v, false, bf.v,
                                                              (short)0, acc[sub], false, false);
        }
        __syncthreads();
    }

    // ---------------- epilogue ----------------
    float*    Cfb = Cf ? Cf + offC : nullptr;
    _Float16* Chb = Ch ? Ch + offC : nullptr;
    #pragma unroll
    for (int sub = 0; sub < NSUB; sub++) {
        #pragma unroll
        for (int r = 0; r < 8; r++) {
            const int gm = m0 + waveM * 16 + r + 8 * hi;
            const int gn = n0 + waveN * (BNT / 2) + sub * 16 + nlane;
            float v = acc[sub][r] * scale;
            if (bias) v += bias[gn];
            if (relu) v = v > 0.f ? v : 0.f;
            if (Cfb) Cfb[(long long)gm * ldc + gn] = v;
            if (Chb) Chb[(long long)gm * ldc + gn] = (_Float16)v;
        }
    }
}

// elementwise fp32 -> f16 (n is a multiple of 2048)
__global__ __launch_bounds__(256) void f32_to_f16_kernel(
    const float* __restrict__ src, _Float16* __restrict__ dst, long long n)
{
    const long long i = ((long long)blockIdx.x * 256 + threadIdx.x) * 8;
    if (i >= n) return;
    float4 a = *(const float4*)(src + i);
    float4 b = *(const float4*)(src + i + 4);
    uint4 o;
    o.x = pack2f(a.x, a.y); o.y = pack2f(a.z, a.w);
    o.z = pack2f(b.x, b.y); o.w = pack2f(b.z, b.w);
    *(uint4*)(dst + i) = o;
}

// Masked softmax over rows of attn[B*H*S, S] in place. One row per block.
__global__ __launch_bounds__(256) void softmax_mask_kernel(
    float* __restrict__ attn, const unsigned char* __restrict__ mask)
{
    __shared__ float red[256];
    const int tid = threadIdx.x;
    const long long row = blockIdx.x;              // b*16384 + h*1024 + qi
    const int qi = (int)(row & 1023);
    const int b  = (int)(row >> 14);
    float* p = attn + row * 1024;
    const unsigned char* mrow = mask + (((long long)b * 1024 + qi) << 10);

    float x[4];
    float mx = -3.4e38f;
    #pragma unroll
    for (int i = 0; i < 4; i++) {
        const int c = tid + i * 256;
        float v = p[c];
        if (mrow[c]) v = -1e9f;
        x[i] = v;
        mx = fmaxf(mx, v);
    }
    red[tid] = mx; __syncthreads();
    for (int s = 128; s > 0; s >>= 1) {
        if (tid < s) red[tid] = fmaxf(red[tid], red[tid + s]);
        __syncthreads();
    }
    mx = red[0]; __syncthreads();

    float sum = 0.f;
    #pragma unroll
    for (int i = 0; i < 4; i++) { x[i] = __expf(x[i] - mx); sum += x[i]; }
    red[tid] = sum; __syncthreads();
    for (int s = 128; s > 0; s >>= 1) {
        if (tid < s) red[tid] += red[tid + s];
        __syncthreads();
    }
    const float inv = 1.0f / red[0];
    #pragma unroll
    for (int i = 0; i < 4; i++) p[tid + i * 256] = x[i] * inv;
}

// outf/outh[row] = LayerNorm(X[row] + R[row]) * g + b    (row length 1024)
__global__ __launch_bounds__(256) void add_ln_kernel(
    const float* __restrict__ X, const float* __restrict__ R,
    const float* __restrict__ g, const float* __restrict__ bta,
    float* __restrict__ outf, _Float16* __restrict__ outh)
{
    __shared__ float red[256];
    const int tid = threadIdx.x;
    const long long row = blockIdx.x;
    const float* xr = X + row * 1024;
    const float* rr = R + row * 1024;

    float v[4];
    float s = 0.f, s2 = 0.f;
    #pragma unroll
    for (int i = 0; i < 4; i++) {
        const int c = tid + i * 256;
        v[i] = xr[c] + rr[c];
        s  += v[i];
        s2 += v[i] * v[i];
    }
    red[tid] = s; __syncthreads();
    for (int t = 128; t > 0; t >>= 1) { if (tid < t) red[tid] += red[tid + t]; __syncthreads(); }
    const float mu = red[0] * (1.0f / 1024.0f); __syncthreads();
    red[tid] = s2; __syncthreads();
    for (int t = 128; t > 0; t >>= 1) { if (tid < t) red[tid] += red[tid + t]; __syncthreads(); }
    const float var  = red[0] * (1.0f / 1024.0f) - mu * mu;
    const float rstd = rsqrtf(var + 1e-5f);
    #pragma unroll
    for (int i = 0; i < 4; i++) {
        const int c = tid + i * 256;
        const float o = (v[i] - mu) * rstd * g[c] + bta[c];
        outf[row * 1024 + c] = o;
        if (outh) outh[row * 1024 + c] = (_Float16)o;
    }
}

extern "C" void kernel_launch(void* const* d_in, const int* in_sizes, int n_in,
                              void* d_out, int out_size, void* d_ws, size_t ws_size,
                              hipStream_t stream) {
    (void)in_sizes; (void)n_in; (void)out_size; (void)ws_size;

    const float* x    = (const float*)d_in[0];
    const unsigned char* mask = (const unsigned char*)d_in[1];
    const float* Wq   = (const float*)d_in[2];
    const float* bq   = (const float*)d_in[3];
    const float* Wk   = (const float*)d_in[4];
    const float* bk   = (const float*)d_in[5];
    const float* Wv   = (const float*)d_in[6];
    const float* bv   = (const float*)d_in[7];
    const float* Wo   = (const float*)d_in[8];
    const float* bo   = (const float*)d_in[9];
    const float* ln1g = (const float*)d_in[10];
    const float* ln1b = (const float*)d_in[11];
    const float* W1   = (const float*)d_in[12];
    const float* b1   = (const float*)d_in[13];
    const float* W2   = (const float*)d_in[14];
    const float* b2   = (const float*)d_in[15];
    const float* ln2g = (const float*)d_in[16];
    const float* ln2b = (const float*)d_in[17];

    float* out      = (float*)d_out;
    float* out_ffn  = out;                          // [8,1024,1024]
    float* out_attn = out + 8388608LL;              // [8,16,1024,1024]

    // ---- workspace layout ----
    _Float16* hw = (_Float16*)d_ws;
    const long long M1 = 1048576LL, M4 = 4194304LL, M8 = 8388608LL;
    _Float16* x16  = hw;                 // 8M halves
    _Float16* wq16 = hw + M8;            // 1M
    _Float16* wk16 = wq16 + M1;
    _Float16* wv16 = wk16 + M1;
    _Float16* wo16 = wv16 + M1;
    _Float16* w116 = wo16 + M1;          // 4M
    _Float16* w216 = w116 + M4;          // 4M
    _Float16* q16  = w216 + M4;          // 8M
    _Float16* k16  = q16 + M8;
    _Float16* v16  = k16 + M8;
    _Float16* c16  = v16 + M8;           // ctx
    _Float16* a16  = c16 + M8;           // attn_out f16
    _Float16* h16  = a16 + M8;           // 32M halves
    float* freg    = (float*)(h16 + 4 * M8);
    float* projf   = freg;               // 8M floats (reused for ffn2 out)
    float* attnf   = freg + M8;          // 8M floats

    const dim3 blk(256);
    const long long HS = 64, BS = 1048576, AH = 1048576, AB = 16777216;

    // 0: one-time fp32 -> f16 conversions (x + all weights)
    f32_to_f16_kernel<<<4096, blk, 0, stream>>>(x,  x16,  M8);
    f32_to_f16_kernel<<< 512, blk, 0, stream>>>(Wq, wq16, M1);
    f32_to_f16_kernel<<< 512, blk, 0, stream>>>(Wk, wk16, M1);
    f32_to_f16_kernel<<< 512, blk, 0, stream>>>(Wv, wv16, M1);
    f32_to_f16_kernel<<< 512, blk, 0, stream>>>(Wo, wo16, M1);
    f32_to_f16_kernel<<<2048, blk, 0, stream>>>(W1, w116, M4);
    f32_to_f16_kernel<<<2048, blk, 0, stream>>>(W2, w216, M4);

    // 1-3: QKV projections -> f16
    gemm_wmma_kernel<_Float16, 128><<<dim3(8, 128, 1), blk, 0, stream>>>(
        x16, 1024, 0, 0, wq16, 1024, 0, 0, 0, nullptr, q16, 1024, 0, 0,
        bq, 1.0f, 0, 1, 8192, 1024, 1024);
    gemm_wmma_kernel<_Float16, 128><<<dim3(8, 128, 1), blk, 0, stream>>>(
        x16, 1024, 0, 0, wk16, 1024, 0, 0, 0, nullptr, k16, 1024, 0, 0,
        bk, 1.0f, 0, 1, 8192, 1024, 1024);
    gemm_wmma_kernel<_Float16, 128><<<dim3(8, 128, 1), blk, 0, stream>>>(
        x16, 1024, 0, 0, wv16, 1024, 0, 0, 0, nullptr, v16, 1024, 0, 0,
        bv, 1.0f, 0, 1, 8192, 1024, 1024);

    // 4: scores = (Q @ K^T)/8 per (b,h) -> f32 straight into attn output region
    gemm_wmma_kernel<_Float16, 128><<<dim3(8, 16, 128), blk, 0, stream>>>(
        q16, 1024, HS, BS, k16, 1024, HS, BS, 1, out_attn, nullptr, 1024, AH, AB,
        nullptr, 0.125f, 0, 16, 1024, 1024, 64);

    // 5: masked softmax in place
    softmax_mask_kernel<<<131072, blk, 0, stream>>>(out_attn, mask);

    // 6: ctx = attn(f32) @ V per (b,h) -> f16 [B,S,H*64]
    gemm_wmma_kernel<float, 64><<<dim3(1, 16, 128), blk, 0, stream>>>(
        out_attn, 1024, AH, AB, v16, 1024, HS, BS, 0, nullptr, c16, 1024, HS, BS,
        nullptr, 1.0f, 0, 16, 1024, 64, 1024);

    // 7: output projection -> f32
    gemm_wmma_kernel<_Float16, 128><<<dim3(8, 128, 1), blk, 0, stream>>>(
        c16, 1024, 0, 0, wo16, 1024, 0, 0, 0, projf, nullptr, 1024, 0, 0,
        bo, 1.0f, 0, 1, 8192, 1024, 1024);

    // 8: attn_out = LN(proj + x) -> f32 + f16
    add_ln_kernel<<<8192, blk, 0, stream>>>(projf, x, ln1g, ln1b, attnf, a16);

    // 9: FFN1 with fused ReLU -> f16 [8192,4096]
    gemm_wmma_kernel<_Float16, 128><<<dim3(32, 128, 1), blk, 0, stream>>>(
        a16, 1024, 0, 0, w116, 4096, 0, 0, 0, nullptr, h16, 4096, 0, 0,
        b1, 1.0f, 1, 1, 8192, 4096, 1024);

    // 10: FFN2 -> f32 (reuse projf)
    gemm_wmma_kernel<_Float16, 128><<<dim3(8, 128, 1), blk, 0, stream>>>(
        h16, 4096, 0, 0, w216, 1024, 0, 0, 0, projf, nullptr, 1024, 0, 0,
        b2, 1.0f, 0, 1, 8192, 1024, 4096);

    // 11: ffn_out = LN(ffn2out + attn_out) -> d_out
    add_ln_kernel<<<8192, blk, 0, stream>>>(projf, attnf, ln2g, ln2b, out_ffn, nullptr);
}
